// TreeCRF_66855460929572
// MI455X (gfx1250) — matrix operand order repaired and checked by reading, not compiled, not run on hardware
//
#include <hip/hip_runtime.h>
#include <math.h>

typedef __attribute__((ext_vector_type(16))) __bf16 v16bf;
typedef __attribute__((ext_vector_type(8)))  __bf16 v8bf;
typedef __attribute__((ext_vector_type(8)))  float  v8f;

#define B_DIM 4096
#define F_DIM 1024
#define C_DIM 2047
#define N_PAD 2048
#define TREE_DEPTH 11

#define BM 128
#define BN 128
#define BK 32
#define LDA 40   // padded LDS row stride (bf16 elems) to spread banks
#define LDB 40

// ---------------- conversion / gather kernels ----------------

__global__ void cvt_x_bf16(const float* __restrict__ x, __bf16* __restrict__ x16, int n) {
  int i = blockIdx.x * blockDim.x + threadIdx.x;
  if (i < n) x16[i] = (__bf16)x[i];
}

__global__ void cvt_w_bf16(const float* __restrict__ w, __bf16* __restrict__ w16) {
  int i = blockIdx.x * blockDim.x + threadIdx.x;   // over N_PAD * F_DIM
  int nrow = i >> 10;                              // F_DIM == 1024
  w16[i] = (nrow < C_DIM) ? (__bf16)w[i] : (__bf16)0.0f;
}

// Compact the (C,C,2,2) pairs tensor down to the C-1 tree edges actually used:
// epe[c*4 + yp*2 + yc] = pairs[parent(c), c, yp, yc]
__global__ void gather_edges(const float* __restrict__ pairs, float* __restrict__ epe) {
  int c = blockIdx.x * blockDim.x + threadIdx.x;
  if (c >= C_DIM) return;
  if (c == 0) { epe[0] = epe[1] = epe[2] = epe[3] = 0.0f; return; }
  int p = (c - 1) >> 1;
  size_t base = ((size_t)p * C_DIM + (size_t)c) * 4;
  epe[c * 4 + 0] = pairs[base + 0];
  epe[c * 4 + 1] = pairs[base + 1];
  epe[c * 4 + 2] = pairs[base + 2];
  epe[c * 4 + 3] = pairs[base + 3];
}

// ---------------- bf16 WMMA GEMM: e = x @ W^T + bias ----------------
// 4 waves / block, each wave owns a 64x64 patch (4x4 WMMA tiles):
// 16 WMMAs per 16 fragment ds_load_b128 per K-step -> ~1 LDS read per WMMA.

__global__ __launch_bounds__(128) void gemm_bf16_wmma(
    const __bf16* __restrict__ x16, const __bf16* __restrict__ w16,
    const float* __restrict__ bias, float* __restrict__ e)
{
  __shared__ __bf16 As[BM * LDA];
  __shared__ __bf16 Bs[BN * LDB];

  const int tid  = threadIdx.x;
  const int m0   = blockIdx.y * BM;
  const int n0   = blockIdx.x * BN;
  const int wave = tid >> 5;
  const int lane = tid & 31;
  const int hf   = lane >> 4;    // lane half-group select
  const int l15  = lane & 15;
  const int waveM = wave >> 1;   // 0..1 -> 64-row patch
  const int waveN = wave & 1;    // 0..1 -> 64-col patch

  // cooperative tile loaders: 128 threads, each moves 4 A + 4 B 16B chunks
  const int ldRow   = tid >> 2;          // 0..31
  const int ldChunk = (tid & 3) * 8;     // bf16 col offset 0/8/16/24

  const __bf16* Aptr = x16 + (size_t)(m0 + ldRow) * F_DIM + ldChunk;
  const __bf16* Bptr = w16 + (size_t)(n0 + ldRow) * F_DIM + ldChunk;

  v8f acc[4][4];
  #pragma unroll
  for (int mi = 0; mi < 4; ++mi)
    #pragma unroll
    for (int ni = 0; ni < 4; ++ni) {
      v8f z = {0.f, 0.f, 0.f, 0.f, 0.f, 0.f, 0.f, 0.f};
      acc[mi][ni] = z;
    }

  union V16 { v16bf v; v8bf h[2]; };

  for (int kt = 0; kt < F_DIM; kt += BK) {
    if (kt + BK < F_DIM) {                       // global_prefetch_b8 next tiles
      __builtin_prefetch(Aptr + kt + BK, 0, 1);
      __builtin_prefetch(Bptr + kt + BK, 0, 1);
    }
    #pragma unroll
    for (int it = 0; it < 4; ++it) {
      *(v8bf*)&As[(ldRow + 32 * it) * LDA + ldChunk] =
          *(const v8bf*)(Aptr + kt + (size_t)(32 * it) * F_DIM);
      *(v8bf*)&Bs[(ldRow + 32 * it) * LDB + ldChunk] =
          *(const v8bf*)(Bptr + kt + (size_t)(32 * it) * F_DIM);
    }
    __syncthreads();

    // A fragment (16x32 bf16): VGPR0-3 hold K = hf*8 + 0..7, VGPR4-7 hold K = 16 + hf*8 + 0..7
    V16 a[4];
    #pragma unroll
    for (int mi = 0; mi < 4; ++mi) {
      int r = (waveM * 64 + mi * 16 + l15) * LDA + hf * 8;
      a[mi].h[0] = *(const v8bf*)&As[r];
      a[mi].h[1] = *(const v8bf*)&As[r + 16];
    }
    // B fragment (32x16 bf16): N = lane&15, VGPR0-3 hold K = hf*8 + 0..7, VGPR4-7 hold K = 16 + hf*8 + 0..7
    V16 b[4];
    #pragma unroll
    for (int ni = 0; ni < 4; ++ni) {
      int r = (waveN * 64 + ni * 16 + l15) * LDB + hf * 8;
      b[ni].h[0] = *(const v8bf*)&Bs[r];
      b[ni].h[1] = *(const v8bf*)&Bs[r + 16];
    }

    #pragma unroll
    for (int mi = 0; mi < 4; ++mi)
      #pragma unroll
      for (int ni = 0; ni < 4; ++ni)
        acc[mi][ni] = __builtin_amdgcn_wmma_f32_16x16x32_bf16(
            false, a[mi].v, false, b[ni].v, (short)0, acc[mi][ni], false, false);
    __syncthreads();
  }

  // epilogue: C/D layout -> VGPR r, lanes: M = hf*8 + r, N = lane&15
  #pragma unroll
  for (int mi = 0; mi < 4; ++mi) {
    #pragma unroll
    for (int ni = 0; ni < 4; ++ni) {
      int gn = n0 + waveN * 64 + ni * 16 + l15;
      if (gn >= C_DIM) continue;
      float bv = bias[gn];
      int gmBase = m0 + waveM * 64 + mi * 16 + hf * 8;
      #pragma unroll
      for (int r = 0; r < 8; ++r)
        e[(size_t)(gmBase + r) * N_PAD + gn] = acc[mi][ni][r] + bv;
    }
  }
}

// ---------------- tree-CRF sum-product, one workgroup per batch row ----------------

__device__ __forceinline__ float lse2(float x, float y) {
  float m = fmaxf(x, y);
  return m + log1pf(__expf(-fabsf(x - y)));
}

__global__ __launch_bounds__(256) void tree_crf(
    const float* __restrict__ e, const float* __restrict__ epe, float* __restrict__ out)
{
  __shared__ float se[N_PAD];
  __shared__ float sa0[N_PAD], sa1[N_PAD], sb0[N_PAD], sb1[N_PAD];
  const int row = blockIdx.x;
  const int tid = threadIdx.x;
  const float* er = e + (size_t)row * N_PAD;

  for (int i = tid; i < C_DIM; i += 256) {
    se[i] = er[i];
    sa0[i] = 0.f; sa1[i] = 0.f; sb0[i] = 0.f; sb1[i] = 0.f;
  }
  __syncthreads();

  // upward (leaves -> root): one thread per parent combines both children
  for (int l = TREE_DEPTH - 1; l >= 1; --l) {
    int h = 1 << (l - 1);
    int pbase = h - 1;
    for (int t = tid; t < h; t += 256) {
      int p = pbase + t;
      float acc0 = 0.f, acc1 = 0.f;
      #pragma unroll
      for (int s = 0; s < 2; ++s) {
        int c = 2 * p + 1 + s;
        float l0 = -se[c] + sa0[c];
        float l1 =  se[c] + sa1[c];
        const float* pe = epe + c * 4;           // [yp][yc]
        acc0 += lse2(pe[0] + l0, pe[1] + l1);
        acc1 += lse2(pe[2] + l0, pe[3] + l1);
      }
      sa0[p] += acc0;
      sa1[p] += acc1;
    }
    __syncthreads();
  }

  // downward (root -> leaves)
  for (int l = 0; l < TREE_DEPTH - 1; ++l) {
    int h = 1 << l;
    int base = h - 1;
    for (int t = tid; t < h; t += 256) {
      int i = base + t;
      float l0 = -se[i] + sb0[i];
      float l1 =  se[i] + sb1[i];
      #pragma unroll
      for (int s = 0; s < 2; ++s) {
        int c = 2 * i + 1 + s;
        const float* pe = epe + c * 4;
        sb0[c] = lse2(l0 + pe[0], l1 + pe[2]);   // sum over y_parent
        sb1[c] = lse2(l0 + pe[1], l1 + pe[3]);
      }
    }
    __syncthreads();
  }

  // softmax over 2 labels == sigmoid of score difference
  for (int i = tid; i < C_DIM; i += 256) {
    float z = 2.f * se[i] + (sa1[i] - sa0[i]) + (sb1[i] - sb0[i]);
    out[(size_t)row * C_DIM + i] = 1.f / (1.f + __expf(-z));
  }
}

// ---------------- launch ----------------

extern "C" void kernel_launch(void* const* d_in, const int* in_sizes, int n_in,
                              void* d_out, int out_size, void* d_ws, size_t ws_size,
                              hipStream_t stream) {
  const float* x     = (const float*)d_in[0];   // (B, F)
  const float* w     = (const float*)d_in[1];   // (C, F)
  const float* bias  = (const float*)d_in[2];   // (C,)
  const float* pairs = (const float*)d_in[3];   // (C, C, 2, 2)
  float* out = (float*)d_out;                   // (B, C)

  // workspace layout
  char* ws = (char*)d_ws;
  __bf16* x16 = (__bf16*)ws;                                               // 8 MiB
  __bf16* w16 = (__bf16*)(ws + (size_t)B_DIM * F_DIM * 2);                 // 4 MiB (padded to 2048 rows)
  float*  e   = (float*)(ws + (size_t)B_DIM * F_DIM * 2
                            + (size_t)N_PAD * F_DIM * 2);                  // 32 MiB (B x N_PAD)
  float*  epe = (float*)((char*)e + (size_t)B_DIM * N_PAD * 4);            // 32 KiB

  cvt_x_bf16<<<(B_DIM * F_DIM) / 256, 256, 0, stream>>>(x, x16, B_DIM * F_DIM);
  cvt_w_bf16<<<(N_PAD * F_DIM) / 256, 256, 0, stream>>>(w, w16);
  gather_edges<<<(C_DIM + 255) / 256, 256, 0, stream>>>(pairs, epe);

  dim3 grid(N_PAD / BN, B_DIM / BM);   // 16 x 32 = 512 blocks
  gemm_bf16_wmma<<<grid, 128, 0, stream>>>(x16, w16, bias, e);

  tree_crf<<<B_DIM, 256, 0, stream>>>(e, epe, out);
}
